// AttentionHead_10625749090339
// MI455X (gfx1250) — matrix operand order repaired and checked by reading, not compiled, not run on hardware
//
#include <hip/hip_runtime.h>
#include <hip/hip_bf16.h>
#include <math.h>

// ---------------- CDNA5 / gfx1250 wave32 WMMA types ----------------
typedef __attribute__((ext_vector_type(16))) __bf16       v16bf;
typedef __attribute__((ext_vector_type(8)))  float        v8f;
typedef __attribute__((ext_vector_type(4)))  unsigned int uint32x4;  // trivial, union-safe
typedef __attribute__((ext_vector_type(4)))  float        floatx4;

#define BM 128
#define BN 128
#define BK 32
#define LDA 40   // (BK + 8) bf16 elements; 80B row stride keeps 16B alignment, avoids bank conflicts
#define LDB 40
#define NTHREADS 256   // 8 wave32 waves per block (WGP fills nicely)

__device__ __forceinline__ unsigned short f32_to_bf16_rne(float f) {
  unsigned int u = __float_as_uint(f);
  u += 0x7FFFu + ((u >> 16) & 1u);
  return (unsigned short)(u >> 16);
}

// ---------- CDNA5 async global->LDS copy (ASYNCcnt-tracked, no VGPR bounce) ----------
// CDNA5 ISA: GLOBAL_LOAD_ASYNC_TO_LDS_B128, GV mode (saddr=off).
// VDST holds the wave-relative LDS byte address = low 32 bits of the generic pointer
// (flat LDS aperture truncates to addr[31:0], ISA §10.2).
__device__ __forceinline__ unsigned lds_off(const void* p) {
  return (unsigned)(unsigned long long)p;
}
__device__ __forceinline__ void async_load_b128(unsigned lds_byte_off, const void* gaddr) {
  asm volatile("global_load_async_to_lds_b128 %0, %1, off"
               :: "v"(lds_byte_off), "v"(gaddr) : "memory");
}
__device__ __forceinline__ void wait_asynccnt0() {
  asm volatile("s_wait_asynccnt 0x0" ::: "memory");
}
__device__ __forceinline__ void wait_asynccnt4() {   // allow 4 newest copies in flight
  asm volatile("s_wait_asynccnt 0x4" ::: "memory");
}

union FragU { uint32x4 q[2]; v16bf v; };      // 32B: one WMMA A/B operand per lane
union Sh16U { uint32x4 q[2]; unsigned short s[16]; };

// One BK=32 slab of wave-level MMA: wave tile 32x64 = 2x4 tiles of 16x16.
// As: [BM][LDA] bf16 row-major (rows = M).  Bt: [BN][LDB] bf16 (rows = N, cols = K).
__device__ __forceinline__ void wave_mma_step(const unsigned short* As,
                                              const unsigned short* Bt,
                                              int m0, int n0, int lane,
                                              v8f acc[2][4]) {
  const int l  = lane & 15;
  const int hi = lane >> 4;   // 0: lanes 0-15, 1: lanes 16-31
  FragU a[2], b[4];
#pragma unroll
  for (int mt = 0; mt < 2; ++mt) {
    // ISA A layout (16-bit, 16x32): lane<16 -> K {0..7, 16..23}; lane>=16 -> K {8..15, 24..31}
    const unsigned short* p = &As[(m0 + mt * 16 + l) * LDA + hi * 8];
    a[mt].q[0] = *(const uint32x4*)(p);
    a[mt].q[1] = *(const uint32x4*)(p + 16);
  }
#pragma unroll
  for (int nt = 0; nt < 4; ++nt) {
    // ISA B layout (32x16): lane<16 -> K {0..7, 16..23} of col l; lane>=16 -> K {8..15, 24..31}
    const unsigned short* p = &Bt[(n0 + nt * 16 + l) * LDB + hi * 8];
    b[nt].q[0] = *(const uint32x4*)(p);
    b[nt].q[1] = *(const uint32x4*)(p + 16);
  }
#pragma unroll
  for (int mt = 0; mt < 2; ++mt)
#pragma unroll
    for (int nt = 0; nt < 4; ++nt)
      acc[mt][nt] = __builtin_amdgcn_wmma_f32_16x16x32_bf16(
          false, a[mt].v, false, b[nt].v, (short)0, acc[mt][nt], false, false);
}

// ================= Kernel 1: QKV projections =================
// Out_bf16[M,N] = X_f32[M,K] @ W_f32[K,N], fp32->bf16 converted on the LDS path.
// blockIdx.z selects which of Q/K/V projection this block computes.
__global__ __launch_bounds__(NTHREADS)
void proj_gemm_f32_bf16(const float* __restrict__ Xk, const float* __restrict__ Xv,
                        const float* __restrict__ Xq,
                        const float* __restrict__ Wq, const float* __restrict__ Wk,
                        const float* __restrict__ Wv,
                        unsigned short* __restrict__ Qo, unsigned short* __restrict__ Ko,
                        unsigned short* __restrict__ Vo,
                        int M, int Kdim, int N) {
  const float* X; const float* W; unsigned short* O;
  if (blockIdx.z == 0)      { X = Xq; W = Wq; O = Qo; }
  else if (blockIdx.z == 1) { X = Xk; W = Wk; O = Ko; }
  else                      { X = Xv; W = Wv; O = Vo; }

  __shared__ unsigned short As[BM * LDA];
  __shared__ unsigned short Bt[BN * LDB];

  const int tid  = threadIdx.x;
  const int lane = tid & 31;
  const int wid  = tid >> 5;
  const int wm0  = (wid & 3) * 32;
  const int wn0  = (wid >> 2) * 64;
  const int gm   = blockIdx.y * BM;
  const int gn   = blockIdx.x * BN;

  const v8f vzero = {0.f, 0.f, 0.f, 0.f, 0.f, 0.f, 0.f, 0.f};
  v8f acc[2][4];
#pragma unroll
  for (int i = 0; i < 2; ++i)
#pragma unroll
    for (int j = 0; j < 4; ++j) acc[i][j] = vzero;

  const int ar  = tid >> 1;            // A tile: 2 threads per row of 32 f32
  const int acb = (tid & 1) * 16;
  const int br  = tid >> 3;            // B tile: 8 threads per row of 128 f32
  const int bcb = (tid & 7) * 16;

  for (int k0 = 0; k0 < Kdim; k0 += BK) {
    {  // A: 128x32 f32 -> bf16 LDS (row-major)
      const float* src = &X[(size_t)(gm + ar) * Kdim + k0 + acb];
      unsigned short* dst = &As[ar * LDA + acb];
      if (k0 + BK < Kdim) __builtin_prefetch(src + BK, 0, 1);  // global_prefetch_b8
#pragma unroll
      for (int i = 0; i < 16; i += 4) {
        floatx4 f = *(const floatx4*)(src + i);
        dst[i + 0] = f32_to_bf16_rne(f.x);
        dst[i + 1] = f32_to_bf16_rne(f.y);
        dst[i + 2] = f32_to_bf16_rne(f.z);
        dst[i + 3] = f32_to_bf16_rne(f.w);
      }
    }
    {  // B: 32x128 f32 -> bf16, stored transposed [n][k] so B-frags are contiguous
      const float* src = &W[(size_t)(k0 + br) * N + gn + bcb];
      if (k0 + BK < Kdim) __builtin_prefetch(src + (size_t)BK * N, 0, 1);
#pragma unroll
      for (int i = 0; i < 16; i += 4) {
        floatx4 f = *(const floatx4*)(src + i);
        Bt[(bcb + i + 0) * LDB + br] = f32_to_bf16_rne(f.x);
        Bt[(bcb + i + 1) * LDB + br] = f32_to_bf16_rne(f.y);
        Bt[(bcb + i + 2) * LDB + br] = f32_to_bf16_rne(f.z);
        Bt[(bcb + i + 3) * LDB + br] = f32_to_bf16_rne(f.w);
      }
    }
    __syncthreads();
    wave_mma_step(As, Bt, wm0, wn0, lane, acc);
    __syncthreads();
  }

  // C layout: VGPR r -> row r (lanes 0-15) / row r+8 (lanes 16-31), col = lane&15
  const int l = lane & 15, hi = lane >> 4;
#pragma unroll
  for (int mt = 0; mt < 2; ++mt)
#pragma unroll
    for (int nt = 0; nt < 4; ++nt)
#pragma unroll
      for (int r = 0; r < 8; ++r) {
        int row = gm + wm0 + mt * 16 + hi * 8 + r;
        int col = gn + wn0 + nt * 16 + l;
        O[(size_t)row * N + col] = f32_to_bf16_rne(acc[mt][nt][r]);
      }
}

// ================= Kernel 2: causal scores = scale * Q @ K^T =================
// Both operands are already bf16 in memory -> pure copies into LDS. Use the CDNA5
// async global->LDS engine with double-buffered LDS: each thread issues 4
// GLOBAL_LOAD_ASYNC_TO_LDS_B128 per K-slab; async loads complete in order, so
// s_wait_asynccnt 0x4 retires the older slab while the next one stays in flight.
// Upper-triangle blocks exit early (causal halves the work); masked elements in
// diagonal blocks are stored but never read by the softmax.
__global__ __launch_bounds__(NTHREADS)
void attn_scores(const unsigned short* __restrict__ Qb,
                 const unsigned short* __restrict__ Kb,
                 float* __restrict__ Sc, int S, int D, float scale) {
  if (blockIdx.x > blockIdx.y) return;  // key-tile entirely above the diagonal

  __shared__ unsigned short As[2][BM * LDA];
  __shared__ unsigned short Bt[2][BN * LDB];

  const int tid  = threadIdx.x;
  const int lane = tid & 31;
  const int wid  = tid >> 5;
  const int wm0  = (wid & 3) * 32;
  const int wn0  = (wid >> 2) * 64;
  const int gm   = blockIdx.y * BM;   // query rows
  const int gn   = blockIdx.x * BN;   // key cols

  const v8f vzero = {0.f, 0.f, 0.f, 0.f, 0.f, 0.f, 0.f, 0.f};
  v8f acc[2][4];
#pragma unroll
  for (int i = 0; i < 2; ++i)
#pragma unroll
    for (int j = 0; j < 4; ++j) acc[i][j] = vzero;

  const int r  = tid >> 1;           // both tiles: 128 rows x 32 cols of bf16
  const int cb = (tid & 1) * 16;
  const unsigned short* qrow = &Qb[(size_t)(gm + r) * D + cb];
  const unsigned short* krow = &Kb[(size_t)(gn + r) * D + cb];

  const int nTiles = D / BK;
  // prologue: slab 0 into buffer 0
  async_load_b128(lds_off(&As[0][r * LDA + cb]),     qrow);
  async_load_b128(lds_off(&As[0][r * LDA + cb + 8]), qrow + 8);
  async_load_b128(lds_off(&Bt[0][r * LDB + cb]),     krow);
  async_load_b128(lds_off(&Bt[0][r * LDB + cb + 8]), krow + 8);

  for (int t = 0; t < nTiles; ++t) {
    const int cur = t & 1;
    if (t + 1 < nTiles) {  // issue next slab into the other buffer
      const int nxt = cur ^ 1;
      const unsigned short* qn = qrow + (size_t)(t + 1) * BK;
      const unsigned short* kn = krow + (size_t)(t + 1) * BK;
      async_load_b128(lds_off(&As[nxt][r * LDA + cb]),     qn);
      async_load_b128(lds_off(&As[nxt][r * LDA + cb + 8]), qn + 8);
      async_load_b128(lds_off(&Bt[nxt][r * LDB + cb]),     kn);
      async_load_b128(lds_off(&Bt[nxt][r * LDB + cb + 8]), kn + 8);
      wait_asynccnt4();   // older slab's 4 copies done; newest 4 still in flight
    } else {
      wait_asynccnt0();
    }
    __syncthreads();      // publish slab t to all waves
    wave_mma_step(As[cur], Bt[cur], wm0, wn0, lane, acc);
    __syncthreads();      // reads of buffer `cur` done before it is re-filled
  }

  const int l = lane & 15, hi = lane >> 4;
#pragma unroll
  for (int mt = 0; mt < 2; ++mt)
#pragma unroll
    for (int nt = 0; nt < 4; ++nt)
#pragma unroll
      for (int c = 0; c < 8; ++c) {
        int row = gm + wm0 + mt * 16 + hi * 8 + c;
        int col = gn + wn0 + nt * 16 + l;
        Sc[(size_t)row * S + col] = acc[mt][nt][c] * scale;
      }
}

// ================= Kernel 3: row softmax over the causal prefix =================
// One block per query row; reads only k <= q, writes bf16 probs zero-filled past
// the diagonal so the P@V GEMM can run dense tiles.
__global__ __launch_bounds__(256)
void softmax_rows(const float* __restrict__ Sc, unsigned short* __restrict__ Pr, int S) {
  const int q   = blockIdx.x;
  const int n   = q + 1;             // causal prefix length
  const int tid = threadIdx.x;
  __shared__ float red[256];

  float m = -3.0e38f;
  for (int k = tid; k < n; k += 256) m = fmaxf(m, Sc[(size_t)q * S + k]);
  red[tid] = m; __syncthreads();
  for (int s = 128; s > 0; s >>= 1) {
    if (tid < s) red[tid] = fmaxf(red[tid], red[tid + s]);
    __syncthreads();
  }
  m = red[0];
  __syncthreads();

  float sum = 0.f;
  for (int k = tid; k < n; k += 256) sum += __expf(Sc[(size_t)q * S + k] - m);
  red[tid] = sum; __syncthreads();
  for (int s = 128; s > 0; s >>= 1) {
    if (tid < s) red[tid] += red[tid + s];
    __syncthreads();
  }
  const float inv = 1.f / red[0];

  for (int k = tid; k < S; k += 256) {
    float p = (k < n) ? __expf(Sc[(size_t)q * S + k] - m) * inv : 0.f;
    Pr[(size_t)q * S + k] = f32_to_bf16_rne(p);
  }
}

// ================= Kernel 4: O = P @ V (bf16 WMMA, f32 out) =================
// K-loop clipped to the query tile's diagonal (probs are zero beyond it).
// A (probs) tile is a pure bf16 copy -> async global->LDS; V needs an element
// scatter for the transposed LDS layout, so it stays on the VGPR path.
__global__ __launch_bounds__(NTHREADS)
void attn_pv(const unsigned short* __restrict__ P,   // [S][S] bf16 probs
             const unsigned short* __restrict__ Vb,  // [S][D] bf16
             float* __restrict__ O, int S, int D) {
  __shared__ unsigned short As[BM * LDA];
  __shared__ unsigned short Bt[BN * LDB];

  const int tid  = threadIdx.x;
  const int lane = tid & 31;
  const int wid  = tid >> 5;
  const int wm0  = (wid & 3) * 32;
  const int wn0  = (wid >> 2) * 64;
  const int gm   = blockIdx.y * BM;   // query rows
  const int gn   = blockIdx.x * BN;   // output dim cols

  const v8f vzero = {0.f, 0.f, 0.f, 0.f, 0.f, 0.f, 0.f, 0.f};
  v8f acc[2][4];
#pragma unroll
  for (int i = 0; i < 2; ++i)
#pragma unroll
    for (int j = 0; j < 4; ++j) acc[i][j] = vzero;

  const int ar  = tid >> 1;            // A (P) tile: 128 rows x 32 keys
  const int acb = (tid & 1) * 16;
  const int br  = tid >> 3;            // B (V) tile: 32 keys x 128 dims
  const int bcb = (tid & 7) * 16;

  const int kend = (gm + BM < S) ? (gm + BM) : S;  // causal: P==0 for k > q
  for (int k0 = 0; k0 < kend; k0 += BK) {
    {  // A: async bf16 copy into LDS (ASYNCcnt path)
      const unsigned short* ps = &P[(size_t)(gm + ar) * S + k0 + acb];
      async_load_b128(lds_off(&As[ar * LDA + acb]),     ps);
      async_load_b128(lds_off(&As[ar * LDA + acb + 8]), ps + 8);
    }
    {  // B: V row-major [key][dim] -> transposed LDS [dim][key]
      Sh16U u;
      const uint32x4* vs = (const uint32x4*)&Vb[(size_t)(k0 + br) * D + gn + bcb];
      u.q[0] = vs[0];
      u.q[1] = vs[1];
#pragma unroll
      for (int i = 0; i < 16; ++i) Bt[(bcb + i) * LDB + br] = u.s[i];
    }
    wait_asynccnt0();
    __syncthreads();
    wave_mma_step(As, Bt, wm0, wn0, lane, acc);
    __syncthreads();
  }

  const int l = lane & 15, hi = lane >> 4;
#pragma unroll
  for (int mt = 0; mt < 2; ++mt)
#pragma unroll
    for (int nt = 0; nt < 4; ++nt)
#pragma unroll
      for (int c = 0; c < 8; ++c) {
        int row = gm + wm0 + mt * 16 + hi * 8 + c;
        int col = gn + wn0 + nt * 16 + l;
        O[(size_t)row * D + col] = acc[mt][nt][c];
      }
}

// ================= host: orchestrate the pipeline =================
extern "C" void kernel_launch(void* const* d_in, const int* in_sizes, int n_in,
                              void* d_out, int out_size, void* d_ws, size_t ws_size,
                              hipStream_t stream) {
  (void)in_sizes; (void)n_in; (void)out_size; (void)ws_size;
  const int Bb = 4, S = 2048, D = 1024;
  const int M = Bb * S;
  const float scale = 1.0f / sqrtf((float)S);  // reference scales by sqrt(Sk)

  // setup_inputs order: Xk, Xv, Xq, Wq, Wk, Wv
  const float* Xk = (const float*)d_in[0];
  const float* Xv = (const float*)d_in[1];
  const float* Xq = (const float*)d_in[2];
  const float* Wq = (const float*)d_in[3];
  const float* Wk = (const float*)d_in[4];
  const float* Wv = (const float*)d_in[5];

  // ws layout (72 MB total):
  //   [0,16M)  Q bf16   [16M,32M) K bf16   [32M,48M) V bf16
  //   [48M,64M) scores f32 (one batch, reused)   [64M,72M) probs bf16 (one batch)
  char* ws = (char*)d_ws;
  unsigned short* Qbf = (unsigned short*)(ws);
  unsigned short* Kbf = (unsigned short*)(ws + ((size_t)16 << 20));
  unsigned short* Vbf = (unsigned short*)(ws + ((size_t)32 << 20));
  float*          Sc  = (float*)         (ws + ((size_t)48 << 20));
  unsigned short* Pr  = (unsigned short*)(ws + ((size_t)64 << 20));

  dim3 blk(NTHREADS);
  proj_gemm_f32_bf16<<<dim3(D / BN, M / BM, 3), blk, 0, stream>>>(
      Xk, Xv, Xq, Wq, Wk, Wv, Qbf, Kbf, Vbf, M, D, D);

  for (int b = 0; b < Bb; ++b) {
    const unsigned short* Qb = Qbf + (size_t)b * S * D;
    const unsigned short* Kb = Kbf + (size_t)b * S * D;
    const unsigned short* Vbp = Vbf + (size_t)b * S * D;
    float* Ob = (float*)d_out + (size_t)b * S * D;
    attn_scores<<<dim3(S / BN, S / BM), blk, 0, stream>>>(Qb, Kb, Sc, S, D, scale);
    softmax_rows<<<dim3(S), dim3(256), 0, stream>>>(Sc, Pr, S);
    attn_pv<<<dim3(D / BN, S / BM), blk, 0, stream>>>(Pr, Vbp, Ob, S, D);
  }
}
